// PointNet2SemSegSSG_64768106823700
// MI455X (gfx1250) — matrix operand order repaired and checked by reading, not compiled, not run on hardware
//
#include <hip/hip_runtime.h>
#include <stdint.h>

typedef __attribute__((ext_vector_type(16))) _Float16 v16h;
typedef __attribute__((ext_vector_type(8)))  _Float16 v8h;
typedef __attribute__((ext_vector_type(8)))  float    v8f;

#define DEV __device__ __forceinline__

// ---------------------------------------------------------------------------
// WMMA fragment helpers (CDNA5 v_wmma_f32_16x16x32_f16 layouts, ISA 7.12.2)
// A (16x32 f16): lane<16 -> row M=lane, K chunks {0..7,16..23};
//                lane>=16 -> row M=lane-16, K chunks {8..15,24..31}
// B (32x16 f16): col N = lane%16, same K chunks as A (one Wt row per element)
// C/D (16x16 f32, 8 VGPR): lane<16 (M=v,N=lane); lane>=16 (M=v+8,N=lane-16)
// ---------------------------------------------------------------------------
DEV v16h cat16(v8h lo, v8h hi) {
  return __builtin_shufflevector(lo, hi, 0,1,2,3,4,5,6,7,8,9,10,11,12,13,14,15);
}
DEV v16h frag16(const _Float16* p, int hi_off) {
  v8h lo = *(const v8h*)p;
  v8h hi = *(const v8h*)(p + hi_off);
  return cat16(lo, hi);
}
DEV float relu_st(float acc, float sv, float tv) {
  float y = acc * sv + tv;
  return y > 0.f ? y : 0.f;
}

// K-loop over one 16-row M tile x NB output tiles: 1 A load feeds NB WMMAs.
// NB is a compile-time constant so no branches surround the WMMAs.
template <int NB>
DEV void gemm_block(const _Float16* abase, int Kpad,
                    const _Float16* wp0, long kstep,
                    v8f acc[NB]) {
  const _Float16* ap = abase;
  const _Float16* wp = wp0;
  for (int kb = 0; kb < Kpad; kb += 32) {
    __builtin_prefetch(wp + kstep, 0, 3);  // next 32 K rows (speculative)
    v16h a = frag16(ap, 16);
    ap += 32;
#pragma unroll
    for (int q = 0; q < NB; q++) {
      v16h bf;
#pragma unroll
      for (int s = 0; s < 16; s++)
        bf[s] = wp[(long)(s + ((s >> 3) << 3)) * (kstep >> 5) + (q << 4)];
      acc[q] = __builtin_amdgcn_wmma_f32_16x16x32_f16(
          false, a, false, bf, (short)0, acc[q], false, false);
    }
    wp += kstep;
  }
}

template <int NB>
DEV void store_relu(const v8f* acc, int nt0, int lane, int rb,
                    const float* s, const float* t, _Float16* Xout, int ldo) {
#pragma unroll
  for (int q = 0; q < NB; q++) {
    int col = ((nt0 + q) << 4) + (lane & 15);
    float sv = s[col], tv = t[col];
#pragma unroll
    for (int v = 0; v < 8; v++)
      Xout[(long)(rb + v) * ldo + col] = (_Float16)relu_st(acc[q][v], sv, tv);
  }
}

template <int NB>
DEV void store_max(const v8f* acc, int nt0, int lane,
                   const float* s, const float* t, float* wmaxrow) {
#pragma unroll
  for (int q = 0; q < NB; q++) {
    int col = ((nt0 + q) << 4) + (lane & 15);
    float sv = s[col], tv = t[col];
    float m = 0.f;  // relu output >= 0, so 0 is a valid max identity
#pragma unroll
    for (int v = 0; v < 8; v++) m = fmaxf(m, relu_st(acc[q][v], sv, tv));
    m = fmaxf(m, __shfl_xor(m, 16, 32));  // fold M=v+8 half (wave32)
    if (lane < 16) wmaxrow[col] = m;
  }
}

// ---------------------------------------------------------------------------
// one MLP layer: Xout = relu((Xin @ W^T)*s + t)
// ntiles is 2 or a multiple of 4 for every layer in this network, so the
// NB specialization is exact and branch-free inside the K loop.
// ---------------------------------------------------------------------------
__device__ void mlp_layer(const _Float16* Xin, int Kpad,
                          const _Float16* W, int Cout,
                          const float* s, const float* t,
                          _Float16* Xout, int ldo,
                          int mtiles, int lane, int wave, int nwaves) {
  int ntiles = Cout >> 4;
  long wrow = (lane & 15) + (long)((lane >> 4) << 3) * Cout;  // B: lane%16 = N col, K rows 8h..
  long kstep = (long)Cout * 32;   // advance 32 K rows
  for (int mt = wave; mt < mtiles; mt += nwaves) {
    int rowbase = mt << 4;
    const _Float16* abase =
        Xin + (long)(rowbase + (lane & 15)) * Kpad + ((lane >> 4) << 3);
    int rb = rowbase + ((lane < 16) ? 0 : 8);
    if ((ntiles & 3) == 0) {
      for (int nt0 = 0; nt0 < ntiles; nt0 += 4) {
        v8f acc[4] = {};
        gemm_block<4>(abase, Kpad, W + wrow + ((long)nt0 << 4), kstep, acc);
        store_relu<4>(acc, nt0, lane, rb, s, t, Xout, ldo);
      }
    } else {  // ntiles == 2 (Cout == 32)
      v8f acc[2] = {};
      gemm_block<2>(abase, Kpad, W + wrow, kstep, acc);
      store_relu<2>(acc, 0, lane, rb, s, t, Xout, ldo);
    }
  }
}

// final MLP layer fused with max-pool over the M (sample) dimension
__device__ void mlp_layer_max(const _Float16* Xin, int Kpad,
                              const _Float16* W, int Cout,
                              const float* s, const float* t,
                              float* wmax,
                              int mtiles, int lane, int wave, int nwaves) {
  int ntiles = Cout >> 4;
  long wrow = (lane & 15) + (long)((lane >> 4) << 3) * Cout;
  long kstep = (long)Cout * 32;
  for (int mt = wave; mt < mtiles; mt += nwaves) {
    int rowbase = mt << 4;
    const _Float16* abase =
        Xin + (long)(rowbase + (lane & 15)) * Kpad + ((lane >> 4) << 3);
    float* wmaxrow = wmax + (long)mt * Cout;
    if ((ntiles & 3) == 0) {
      for (int nt0 = 0; nt0 < ntiles; nt0 += 4) {
        v8f acc[4] = {};
        gemm_block<4>(abase, Kpad, W + wrow + ((long)nt0 << 4), kstep, acc);
        store_max<4>(acc, nt0, lane, s, t, wmaxrow);
      }
    } else {
      v8f acc[2] = {};
      gemm_block<2>(abase, Kpad, W + wrow, kstep, acc);
      store_max<2>(acc, 0, lane, s, t, wmaxrow);
    }
  }
}

// ---------------------------------------------------------------------------
// Weight preprocessing: W f32 [N][K] -> Wt f16 [Kpad][Npad] (zero padded),
// batched over blockIdx.y (for per-centroid refine weight stacks)
// ---------------------------------------------------------------------------
__global__ void k_prep_w(const float* __restrict__ W, int N, int K,
                         _Float16* __restrict__ Wt, int Npad, int Kpad) {
  long j = blockIdx.y;
  W  += j * (long)N * K;
  Wt += j * (long)Npad * Kpad;
  long total = (long)Kpad * Npad;
  for (long idx = (long)blockIdx.x * blockDim.x + threadIdx.x; idx < total;
       idx += (long)gridDim.x * blockDim.x) {
    long k = idx / Npad, n = idx - k * Npad;
    float v = (k < K && n < N) ? W[n * (long)K + k] : 0.f;
    Wt[idx] = (_Float16)v;
  }
}

// ---------------------------------------------------------------------------
// Farthest point sampling: one workgroup per batch; sequential argmax rounds
// ---------------------------------------------------------------------------
__global__ void k_fps(const float* __restrict__ xyz, int stride, int N, int npoint,
                      int* __restrict__ fidx, float* __restrict__ new_xyz) {
  int b = blockIdx.x, tid = threadIdx.x;
  const int T = 256;
  __shared__ float dist[1024];
  __shared__ float rv[256];
  __shared__ int   ri[256];
  __shared__ float cur[3];
  const float* base = xyz + (long)b * N * stride;
  for (int i = tid; i < N; i += T) dist[i] = 1e10f;
  if (tid == 0) {
    fidx[(long)b * npoint] = 0;
    for (int c = 0; c < 3; c++) { cur[c] = base[c]; new_xyz[((long)b * npoint) * 3 + c] = base[c]; }
  }
  __syncthreads();
  for (int k = 1; k < npoint; k++) {
    float bx = cur[0], by = cur[1], bz = cur[2];
    float best = -1.f; int bi = 0;
    for (int i = tid; i < N; i += T) {
      float dx = base[(long)i * stride]     - bx;
      float dy = base[(long)i * stride + 1] - by;
      float dz = base[(long)i * stride + 2] - bz;
      float d = dx * dx + dy * dy + dz * dz;
      float m = fminf(dist[i], d);
      dist[i] = m;
      if (m > best || (m == best && i < bi)) { best = m; bi = i; }
    }
    rv[tid] = best; ri[tid] = bi;
    __syncthreads();
    for (int s = 128; s > 0; s >>= 1) {
      if (tid < s) {
        if (rv[tid + s] > rv[tid] || (rv[tid + s] == rv[tid] && ri[tid + s] < ri[tid])) {
          rv[tid] = rv[tid + s]; ri[tid] = ri[tid + s];
        }
      }
      __syncthreads();
    }
    if (tid == 0) {
      int w = ri[0];
      fidx[(long)b * npoint + k] = w;
      for (int c = 0; c < 3; c++) {
        float v = base[(long)w * stride + c];
        cur[c] = v; new_xyz[((long)b * npoint + k) * 3 + c] = v;
      }
    }
    __syncthreads();
  }
}

// ---------------------------------------------------------------------------
// Ball query: first nsample indices within radius (pad with first, else 0)
// ---------------------------------------------------------------------------
__global__ void k_ballquery(const float* __restrict__ xyz, int xstride, int Nsrc,
                            const float* __restrict__ new_xyz, int npoint, int total,
                            float r2, int ns, int* __restrict__ gidx) {
  int t = blockIdx.x * blockDim.x + threadIdx.x;
  if (t >= total) return;
  int b = t / npoint;
  const float* nx = new_xyz + (long)t * 3;
  float cx = nx[0], cy = nx[1], cz = nx[2];
  int* out = gidx + (long)t * ns;
  int cnt = 0, first = 0;
  const float* base = xyz + (long)b * Nsrc * xstride;
  for (int j = 0; j < Nsrc; j++) {
    float dx = base[(long)j * xstride]     - cx;
    float dy = base[(long)j * xstride + 1] - cy;
    float dz = base[(long)j * xstride + 2] - cz;
    float d2 = dx * dx + dy * dy + dz * dz;
    if (d2 < r2) {
      if (cnt == 0) first = j;
      out[cnt++] = j;
      if (cnt == ns) break;
    }
  }
  for (int k = cnt; k < ns; k++) out[k] = (cnt == 0) ? 0 : first;
}

// identity group indices + zero centers (for global-SA / refine reuse)
__global__ void k_init_aux(int* __restrict__ ident, float* __restrict__ zeros,
                           int nident, int nzero) {
  int t = blockIdx.x * blockDim.x + threadIdx.x;
  if (t < nident) ident[t] = t & 63;
  if (t < nzero)  zeros[t] = 0.f;
}

// ---------------------------------------------------------------------------
// Fused SA stage: one workgroup (4 waves) per group.
// Gathers [ns x K0] into LDS, runs 3 WMMA MLP layers in LDS, max-pools.
// perGroup!=0 -> weights/scale/bias indexed by group id (refine's 21 MLPs).
// ---------------------------------------------------------------------------
__global__ void k_sa_fused(
    const float* __restrict__ xyz, int xstride,
    const float* __restrict__ feats, int fstride, int Cf, int Nsrc,
    const float* __restrict__ new_xyz, const int* __restrict__ gidx,
    int npoint, int ns, int K0pad,
    const _Float16* __restrict__ W0, const float* __restrict__ s0, const float* __restrict__ t0, int C1,
    const _Float16* __restrict__ W1, const float* __restrict__ s1, const float* __restrict__ t1, int C2,
    const _Float16* __restrict__ W2, const float* __restrict__ s2, const float* __restrict__ t2, int C3,
    int perGroup,
    float* __restrict__ outf32, _Float16* __restrict__ outf16, int ld16) {
  extern __shared__ char smem[];
  int gid = blockIdx.x;
  int b = gid / npoint;
  int i = gid - b * npoint;
  int tid = threadIdx.x, lane = tid & 31, wave = tid >> 5, nwaves = blockDim.x >> 5;
  int mtiles = ns >> 4;

  _Float16* A  = (_Float16*)smem;
  _Float16* X1 = A  + (long)ns * K0pad;
  _Float16* X2 = X1 + (long)ns * C1;
  float*  wmax = (float*)(X2 + (long)ns * C2);

  const _Float16* w0 = W0; const float* s0p = s0; const float* t0p = t0;
  const _Float16* w1 = W1; const float* s1p = s1; const float* t1p = t1;
  const _Float16* w2 = W2; const float* s2p = s2; const float* t2p = t2;
  if (perGroup) {
    w0 += (long)i * K0pad * C1; s0p += (long)i * C1; t0p += (long)i * C1;
    w1 += (long)i * C1 * C2;    s1p += (long)i * C2; t1p += (long)i * C2;
    w2 += (long)i * C2 * C3;    s2p += (long)i * C3; t2p += (long)i * C3;
  }

  // gather + center + f16 convert into LDS (zero padded to K0pad)
  const int* gi = gidx + (long)gid * ns;
  const float* nx = new_xyz + (long)gid * 3;
  int total = ns * K0pad;
  for (int idx = tid; idx < total; idx += blockDim.x) {
    int r = idx / K0pad, c = idx - r * K0pad;
    float v = 0.f;
    int g = gi[r];
    long src = (long)b * Nsrc + g;
    if (c < 3)          v = xyz[src * xstride + c] - nx[c];
    else if (c < 3 + Cf) v = feats[src * fstride + (c - 3)];
    A[idx] = (_Float16)v;
  }
  __syncthreads();
  mlp_layer(A,  K0pad, w0, C1, s0p, t0p, X1, C1, mtiles, lane, wave, nwaves);
  __syncthreads();
  mlp_layer(X1, C1,    w1, C2, s1p, t1p, X2, C2, mtiles, lane, wave, nwaves);
  __syncthreads();
  mlp_layer_max(X2, C2, w2, C3, s2p, t2p, wmax, mtiles, lane, wave, nwaves);
  __syncthreads();
  for (int c = tid; c < C3; c += blockDim.x) {
    float m = 0.f;
    for (int mt = 0; mt < mtiles; mt++) m = fmaxf(m, wmax[(long)mt * C3 + c]);
    if (outf32) outf32[(long)gid * C3 + c]  = m;
    if (outf16) outf16[(long)gid * ld16 + c] = (_Float16)m;
  }
}

// ---------------------------------------------------------------------------
// Generic WMMA GEMM for FC heads: one 16x16 tile per wave, fused epilogue
// relu==1: y = relu(acc*s+t); relu==0: y = acc + bias (padded cols -> bias 0)
// ---------------------------------------------------------------------------
__global__ void k_gemm(const _Float16* __restrict__ A, int Kpad, int M,
                       const _Float16* __restrict__ W, int Npad,
                       const float* __restrict__ s, const float* __restrict__ t,
                       const float* __restrict__ bias, int Nout,
                       float* __restrict__ o32, int ld32,
                       _Float16* __restrict__ o16, int ld16, int relu) {
  int tid = threadIdx.x, lane = tid & 31, wave = tid >> 5;
  int mt_count = M >> 4, nt_count = Npad >> 4;
  int tile = blockIdx.x * (blockDim.x >> 5) + wave;
  if (tile >= mt_count * nt_count) return;
  int mt = tile / nt_count, nt = tile - mt * nt_count;
  v8f acc[1] = {};
  const _Float16* abase =
      A + (long)((mt << 4) + (lane & 15)) * Kpad + ((lane >> 4) << 3);
  gemm_block<1>(abase, Kpad, W + (lane & 15) + (long)((lane >> 4) << 3) * Npad + (nt << 4), (long)Npad * 32, acc);
  int col = (nt << 4) + (lane & 15);
  int rb = (mt << 4) + ((lane < 16) ? 0 : 8);
  if (relu) {
    float sv = s[col], tv = t[col];
#pragma unroll
    for (int v = 0; v < 8; v++) {
      float y = relu_st(acc[0][v], sv, tv);
      if (o32) o32[(long)(rb + v) * ld32 + col] = y;
      if (o16) o16[(long)(rb + v) * ld16 + col] = (_Float16)y;
    }
  } else {
    float bv = (col < Nout) ? bias[col] : 0.f;
#pragma unroll
    for (int v = 0; v < 8; v++) {
      float y = acc[0][v] + bv;
      if (o32) o32[(long)(rb + v) * ld32 + col] = y;
      if (o16) o16[(long)(rb + v) * ld16 + col] = (_Float16)y;
    }
  }
}

// ---------------------------------------------------------------------------
// 32 nearest neighbors of each estimated centroid: one wave per (b, j)
// ---------------------------------------------------------------------------
__global__ void k_topk(const float* __restrict__ pc, const float* __restrict__ est,
                       int* __restrict__ nnidx) {
  int g = blockIdx.x;
  int b = g / 21, j = g - b * 21;
  int lane = threadIdx.x & 31;
  float cx = est[b * 64 + j * 3 + 0];
  float cy = est[b * 64 + j * 3 + 1];
  float cz = est[b * 64 + j * 3 + 2];
  float d[32];
#pragma unroll
  for (int tt = 0; tt < 32; tt++) {
    int p = lane + (tt << 5);
    const float* q = pc + ((long)b * 1024 + p) * 6;
    float dx = q[0] - cx, dy = q[1] - cy, dz = q[2] - cz;
    d[tt] = dx * dx + dy * dy + dz * dz;
  }
  unsigned mask = 0u;
  for (int it = 0; it < 32; it++) {
    float v = 3.4e38f; int li = 0; bool has = false;
#pragma unroll
    for (int tt = 0; tt < 32; tt++) {
      if (!((mask >> tt) & 1u) && (!has || d[tt] < v)) { v = d[tt]; li = tt; has = true; }
    }
    int gcand = lane + (li << 5);
    if (!has) v = 3.4e38f;
    for (int off = 16; off; off >>= 1) {
      float ov = __shfl_xor(v, off, 32);
      int   og = __shfl_xor(gcand, off, 32);
      if (ov < v || (ov == v && og < gcand)) { v = ov; gcand = og; }
    }
    if ((gcand & 31) == lane) mask |= 1u << (gcand >> 5);
    if (lane == 0) nnidx[(long)g * 32 + it] = gcand;
  }
}

// featmap cols 0..2 = est centroid, 515..543 = zero pad (refine wrote 3..514)
__global__ void k_featmap_fix(const float* __restrict__ est, _Float16* __restrict__ fm) {
  int t = blockIdx.x * blockDim.x + threadIdx.x;
  if (t >= 672) return;
  int b = t / 21, j = t - b * 21;
  for (int c = 0; c < 3; c++) fm[(long)t * 544 + c] = (_Float16)est[b * 64 + j * 3 + c];
  for (int c = 515; c < 544; c++) fm[(long)t * 544 + c] = (_Float16)0.f;
}

__global__ void k_store_est(const float* __restrict__ est, float* __restrict__ out) {
  int t = blockIdx.x * blockDim.x + threadIdx.x;
  if (t >= 32 * 63) return;
  int b = t / 63, r = t - b * 63;
  out[t] = est[b * 64 + r];
}

__global__ void k_store_refine(const float* __restrict__ f2o, float* __restrict__ out) {
  int t = blockIdx.x * blockDim.x + threadIdx.x;
  if (t >= 2016) return;
  out[2016 + t] = f2o[(t / 3) * 16 + (t % 3)];
}

// ---------------------------------------------------------------------------
// Host orchestration
// Input order (insertion order of setup_inputs dict, recursively):
// 0:pointcloud; 1..36: sa[g][l] {W,s,t}; 37..42: fc.hidden {W,s,t}x2;
// 43,44: fc.out {W,b}; 45..53: refine[0..2] {W,s,t}; 54..62: fc2.hidden x3;
// 63,64: fc2.out {W,b}
// ---------------------------------------------------------------------------
extern "C" void kernel_launch(void* const* d_in, const int* in_sizes, int n_in,
                              void* d_out, int out_size, void* d_ws, size_t ws_size,
                              hipStream_t stream) {
  (void)in_sizes; (void)n_in; (void)out_size; (void)ws_size;
  const float* pc = (const float*)d_in[0];
  auto F = [&](int i) { return (const float*)d_in[i]; };

  char* wsp = (char*)d_ws;
  size_t off = 0;
  auto alloc = [&](size_t bytes) -> void* {
    off = (off + 255) & ~(size_t)255;
    void* p = wsp + off;
    off += bytes;
    return p;
  };
  auto prep = [&](int wi, int N, int K, int batch) -> _Float16* {
    int Kpad = (K + 31) & ~31, Npad = (N + 15) & ~15;
    _Float16* wt = (_Float16*)alloc((size_t)batch * Kpad * Npad * sizeof(_Float16));
    long elems = (long)Kpad * Npad;
    int blocks = (int)((elems + 255) / 256);
    if (blocks > 2048) blocks = 2048;
    k_prep_w<<<dim3(blocks, batch), dim3(256), 0, stream>>>(F(wi), N, K, wt, Npad, Kpad);
    return wt;
  };

  static const int sa_ci[4][3] = {{6,32,32},{67,64,64},{131,128,128},{259,256,512}};
  static const int sa_co[4][3] = {{32,32,64},{64,64,128},{128,128,256},{256,512,1024}};
  _Float16* Wsa[4][3];
  for (int g = 0; g < 4; g++)
    for (int l = 0; l < 3; l++)
      Wsa[g][l] = prep(1 + (g * 3 + l) * 3, sa_co[g][l], sa_ci[g][l], 1);
  _Float16* fcW0 = prep(37, 512, 1024, 1);
  _Float16* fcW1 = prep(40, 128, 512, 1);
  _Float16* fcWo = prep(43, 63, 128, 1);
  _Float16* rW0  = prep(45, 64, 6, 21);
  _Float16* rW1  = prep(48, 128, 64, 21);
  _Float16* rW2  = prep(51, 512, 128, 21);
  _Float16* f2W0 = prep(54, 256, 515, 1);
  _Float16* f2W1 = prep(57, 128, 256, 1);
  _Float16* f2W2 = prep(60, 64, 128, 1);
  _Float16* f2Wo = prep(63, 3, 64, 1);

  int*   fidx1 = (int*)alloc(32 * 512 * 4);
  float* nx1   = (float*)alloc(32 * 512 * 3 * 4);
  int*   gid1  = (int*)alloc((size_t)32 * 512 * 64 * 4);
  float* feat1 = (float*)alloc((size_t)32 * 512 * 64 * 4);
  int*   fidx2 = (int*)alloc(32 * 256 * 4);
  float* nx2   = (float*)alloc(32 * 256 * 3 * 4);
  int*   gid2  = (int*)alloc((size_t)32 * 256 * 64 * 4);
  float* feat2 = (float*)alloc((size_t)32 * 256 * 128 * 4);
  int*   fidx3 = (int*)alloc(32 * 64 * 4);
  float* nx3   = (float*)alloc(32 * 64 * 3 * 4);
  int*   gid3  = (int*)alloc((size_t)32 * 64 * 64 * 4);
  float* feat3 = (float*)alloc((size_t)32 * 64 * 256 * 4);
  int*   ident = (int*)alloc(32 * 64 * 4);
  float* zeros3 = (float*)alloc(672 * 3 * 4);
  _Float16* gfeat16 = (_Float16*)alloc((size_t)32 * 1024 * 2);
  _Float16* fch0 = (_Float16*)alloc((size_t)32 * 512 * 2);
  _Float16* fch1 = (_Float16*)alloc((size_t)32 * 128 * 2);
  float* estbuf  = (float*)alloc(32 * 64 * 4);
  int*   nnidx   = (int*)alloc(672 * 32 * 4);
  _Float16* featmap = (_Float16*)alloc((size_t)672 * 544 * 2);
  _Float16* f2a = (_Float16*)alloc((size_t)672 * 256 * 2);
  _Float16* f2b = (_Float16*)alloc((size_t)672 * 128 * 2);
  _Float16* f2c = (_Float16*)alloc((size_t)672 * 64 * 2);
  float* f2o = (float*)alloc((size_t)672 * 16 * 4);

  k_init_aux<<<dim3(8), dim3(256), 0, stream>>>(ident, zeros3, 32 * 64, 672 * 3);

  auto sa = [&](const float* xyz, int xstride, const float* feats, int fstride, int Cf, int Nsrc,
                const float* nxyz, const int* gidx, int npoint, int ns,
                _Float16* w0, const float* s0, const float* t0, int C1,
                _Float16* w1, const float* s1, const float* t1, int C2,
                _Float16* w2, const float* s2, const float* t2, int C3,
                int perGroup, float* o32, _Float16* o16, int ld16) {
    int K0 = 3 + Cf, K0pad = (K0 + 31) & ~31;
    int mtiles = ns / 16;
    size_t lds = (size_t)ns * K0pad * 2 + (size_t)ns * C1 * 2 + (size_t)ns * C2 * 2 +
                 (size_t)mtiles * C3 * 4;
    k_sa_fused<<<dim3(32 * npoint), dim3(128), lds, stream>>>(
        xyz, xstride, feats, fstride, Cf, Nsrc, nxyz, gidx, npoint, ns, K0pad,
        w0, s0, t0, C1, w1, s1, t1, C2, w2, s2, t2, C3, perGroup, o32, o16, ld16);
  };
  auto gemm = [&](const _Float16* A, int Kpad, int M, const _Float16* W, int Npad,
                  const float* s, const float* t, const float* bias, int Nout,
                  float* o32, int ld32, _Float16* o16, int ld16, int relu) {
    int tiles = (M / 16) * (Npad / 16);
    int blocks = (tiles + 3) / 4;
    k_gemm<<<dim3(blocks), dim3(128), 0, stream>>>(A, Kpad, M, W, Npad, s, t, bias, Nout,
                                                   o32, ld32, o16, ld16, relu);
  };

  // ---- SA stage 1: 1024 -> 512 groups, r=0.1, MLP 6->32->32->64
  k_fps<<<dim3(32), dim3(256), 0, stream>>>(pc, 6, 1024, 512, fidx1, nx1);
  k_ballquery<<<dim3((32 * 512 + 255) / 256), dim3(256), 0, stream>>>(
      pc, 6, 1024, nx1, 512, 32 * 512, 0.01f, 64, gid1);
  sa(pc, 6, pc + 3, 6, 3, 1024, nx1, gid1, 512, 64,
     Wsa[0][0], F(2), F(3), 32, Wsa[0][1], F(5), F(6), 32, Wsa[0][2], F(8), F(9), 64,
     0, feat1, nullptr, 0);

  // ---- SA stage 2: 512 -> 256, r=0.2, MLP 67->64->64->128
  k_fps<<<dim3(32), dim3(256), 0, stream>>>(nx1, 3, 512, 256, fidx2, nx2);
  k_ballquery<<<dim3((32 * 256 + 255) / 256), dim3(256), 0, stream>>>(
      nx1, 3, 512, nx2, 256, 32 * 256, 0.04f, 64, gid2);
  sa(nx1, 3, feat1, 64, 64, 512, nx2, gid2, 256, 64,
     Wsa[1][0], F(11), F(12), 64, Wsa[1][1], F(14), F(15), 64, Wsa[1][2], F(17), F(18), 128,
     0, feat2, nullptr, 0);

  // ---- SA stage 3: 256 -> 64, r=0.4, MLP 131->128->128->256
  k_fps<<<dim3(32), dim3(256), 0, stream>>>(nx2, 3, 256, 64, fidx3, nx3);
  k_ballquery<<<dim3((32 * 64 + 255) / 256), dim3(256), 0, stream>>>(
      nx2, 3, 256, nx3, 64, 32 * 64, 0.16f, 64, gid3);
  sa(nx2, 3, feat2, 128, 128, 256, nx3, gid3, 64, 64,
     Wsa[2][0], F(20), F(21), 128, Wsa[2][1], F(23), F(24), 128, Wsa[2][2], F(26), F(27), 256,
     0, feat3, nullptr, 0);

  // ---- Global SA (identity grouping, zero centers): MLP 259->256->512->1024
  sa(nx3, 3, feat3, 256, 256, 64, zeros3, ident, 1, 64,
     Wsa[3][0], F(29), F(30), 256, Wsa[3][1], F(32), F(33), 512, Wsa[3][2], F(35), F(36), 1024,
     0, nullptr, gfeat16, 1024);

  // ---- FC head: 1024 -> 512 -> 128 -> 63 (est centroids)
  gemm(gfeat16, 1024, 32, fcW0, 512, F(38), F(39), nullptr, 512, nullptr, 0, fch0, 512, 1);
  gemm(fch0, 512, 32, fcW1, 128, F(41), F(42), nullptr, 128, nullptr, 0, fch1, 128, 1);
  gemm(fch1, 128, 32, fcWo, 64, nullptr, nullptr, F(44), 63, estbuf, 64, nullptr, 0, 0);
  k_store_est<<<dim3((2016 + 255) / 256), dim3(256), 0, stream>>>(estbuf, (float*)d_out);

  // ---- 32-NN per centroid + per-centroid refine MLPs 6->64->128->512 (max)
  k_topk<<<dim3(672), dim3(32), 0, stream>>>(pc, estbuf, nnidx);
  sa(pc, 6, pc + 3, 6, 3, 1024, zeros3, nnidx, 21, 32,
     rW0, F(46), F(47), 64, rW1, F(49), F(50), 128, rW2, F(52), F(53), 512,
     1, nullptr, featmap + 3, 544);
  k_featmap_fix<<<dim3(3), dim3(256), 0, stream>>>(estbuf, featmap);

  // ---- FC2 head: 515 -> 256 -> 128 -> 64 -> 3
  gemm(featmap, 544, 672, f2W0, 256, F(55), F(56), nullptr, 256, nullptr, 0, f2a, 256, 1);
  gemm(f2a, 256, 672, f2W1, 128, F(58), F(59), nullptr, 128, nullptr, 0, f2b, 128, 1);
  gemm(f2b, 128, 672, f2W2, 64, F(61), F(62), nullptr, 64, nullptr, 0, f2c, 64, 1);
  gemm(f2c, 64, 672, f2Wo, 16, nullptr, nullptr, F(64), 3, f2o, 16, nullptr, 0, 0);
  k_store_refine<<<dim3((2016 + 255) / 256), dim3(256), 0, stream>>>(f2o, (float*)d_out);
}
